// MultiHeadAttention_42013370090260
// MI455X (gfx1250) — compile-verified, matrix operand order checked by reading, not run on hardware
//
#include <hip/hip_runtime.h>

// ---------------------------------------------------------------------------
// MHA forward for MI455X (gfx1250, wave32, WMMA bf16 path)
//   B=2, T=2048, E=1024, H=16, hd=64, N3=3072
// Prep:     x -> bf16 (xb),  W -> bf16 transposed (Wt[N][K])
// Kernel 1: qkv = x@W + b. Pure global-b128 + WMMA (A and B fragments are both
//           two aligned 16B loads in native WMMA layout; weights L2-resident).
// Kernel 2: per-wave flash attention (online softmax), causal, scale sqrt(3)/32
//           applied after masking (reference quirk), fp32 output.
// ---------------------------------------------------------------------------

typedef __attribute__((ext_vector_type(16))) __bf16 v16bf;
typedef __attribute__((ext_vector_type(8)))  __bf16 v8bf;
typedef __attribute__((ext_vector_type(8)))  float  v8f;

#define T_SEQ   2048
#define E_DIM   1024
#define N3      3072
#define HEADS   16
#define HD      64
#define BH      32            // B * HEADS
#define M_TOT   (2 * T_SEQ)   // 4096 rows of x
#define SCALE   0.05412658773652741f   // 1/sqrt(1024/3) = sqrt(3)/32
#define LDSTRIDE 40           // bf16 elements per LDS row: 80 B = 5*16 B

static __device__ __forceinline__ v8f wmma_bf16(v16bf a, v16bf b, v8f c) {
    return __builtin_amdgcn_wmma_f32_16x16x32_bf16(
        /*neg_a=*/false, a, /*neg_b=*/false, b,
        /*c_mod=*/(short)0, c, /*reuse_a=*/false, /*reuse_b=*/false);
}

// Fragment loader: elements 0..7 at p[0..7], elements 8..15 at p[16..23]
// (two aligned 16-byte chunks at +0 and +32 bytes) -- the 16-bit A/B VGPR
// layout for a lane, given p = row_base + half*8.
static __device__ __forceinline__ v16bf load_frag(const __bf16* p) {
    v8bf lo = *(const v8bf*)(p);
    v8bf hi = *(const v8bf*)(p + 16);
    return __builtin_shufflevector(lo, hi, 0, 1, 2, 3, 4, 5, 6, 7,
                                           8, 9, 10, 11, 12, 13, 14, 15);
}

#define LDS_FENCE() asm volatile("" ::: "memory")

// ---------------------------------------------------------------------------
// Prep A: x (fp32) -> xb (bf16), 4M elements, 8 per thread.
// ---------------------------------------------------------------------------
__global__ __launch_bounds__(256)
void cvt_x_bf16(const float* __restrict__ x, __bf16* __restrict__ xb) {
    const int i = (blockIdx.x * 256 + threadIdx.x) * 8;
    const float* p = x + i;
    v8bf o;
    #pragma unroll
    for (int j = 0; j < 8; ++j) o[j] = (__bf16)p[j];
    *(v8bf*)(xb + i) = o;
}

// ---------------------------------------------------------------------------
// Prep B: W[K][N] fp32 -> Wt[N][K] bf16 via 32x32 LDS tile transpose.
// grid=(N3/32, E_DIM/32), block=256.
// ---------------------------------------------------------------------------
__global__ __launch_bounds__(256)
void transpose_w_bf16(const float* __restrict__ W, __bf16* __restrict__ Wt) {
    __shared__ __bf16 sT[32][33];
    const int t  = threadIdx.x;
    const int r  = t >> 3;           // 0..31
    const int c4 = (t & 7) * 4;      // 0,4,...,28
    const int n0 = blockIdx.x * 32;
    const int k0 = blockIdx.y * 32;

    const float* wp = W + (k0 + r) * N3 + n0 + c4;   // coalesced fp32 reads
    #pragma unroll
    for (int i = 0; i < 4; ++i) sT[c4 + i][r] = (__bf16)wp[i];
    __syncthreads();

    __bf16* op = Wt + (n0 + r) * E_DIM + k0 + c4;    // coalesced bf16 writes
    #pragma unroll
    for (int i = 0; i < 4; ++i) op[i] = sT[r][c4 + i];
}

// ---------------------------------------------------------------------------
// Kernel 1: QKV projection GEMM.  grid=(48,32), block=128 (4 waves).
// Wave w: rows [blockIdx.y*128 + w*32, +32), cols [blockIdx.x*64, +64).
// No LDS, no barriers: all fragments are 2x global_load_b128 from L2.
// ---------------------------------------------------------------------------
__global__ __launch_bounds__(128)
void mha_qkv_gemm(const __bf16* __restrict__ xb, const __bf16* __restrict__ Wt,
                  const float* __restrict__ bias,
                  __bf16* __restrict__ Qb, __bf16* __restrict__ Kb,
                  __bf16* __restrict__ Vt) {
    const int lane = threadIdx.x & 31;
    const int wave = threadIdx.x >> 5;
    const int half = lane >> 4;
    const int mrow = lane & 15;
    const int base = half * 8;

    const int M0 = blockIdx.y * 128 + wave * 32;
    const int N0 = blockIdx.x * 64;

    const __bf16* arow0 = xb + (M0 + mrow) * E_DIM + base;
    const __bf16* arow1 = arow0 + 16 * E_DIM;
    const __bf16* brow  = Wt + (N0 + mrow) * E_DIM + base;  // + j*16*E_DIM

    v8f acc[2][4] = {};             // 32(M) x 64(N) fp32 accumulators

    #pragma unroll 2
    for (int kk = 0; kk < E_DIM; kk += 32) {
        const v16bf a0 = load_frag(arow0 + kk);
        const v16bf a1 = load_frag(arow1 + kk);
        #pragma unroll
        for (int j = 0; j < 4; ++j) {
            const v16bf bf = load_frag(brow + j * 16 * E_DIM + kk);
            acc[0][j] = wmma_bf16(a0, bf, acc[0][j]);
            acc[1][j] = wmma_bf16(a1, bf, acc[1][j]);
        }
    }

    // ---- epilogue: sel/h/bidx are workgroup-uniform; one scalar branch ----
    const int sel  = N0 >> 10;            // 0=Q 1=K 2=V
    const int h    = (N0 & 1023) >> 6;    // head (uniform: 64-col tile)
    const int bidx = M0 >> 11;            // batch (uniform: 32-row tile)
    const int bh   = bidx * HEADS + h;
    const int t0b  = (M0 & (T_SEQ - 1)) + half * 8;  // + s*16 + e

    if (sel == 2) {
        // V^T [BH][64][T]: 8 e-values contiguous in t -> one 16B store each
        #pragma unroll
        for (int j = 0; j < 4; ++j) {
            const float bv = bias[N0 + j * 16 + mrow];
            const int d = j * 16 + mrow;
            #pragma unroll
            for (int s = 0; s < 2; ++s) {
                v8bf pk;
                #pragma unroll
                for (int e = 0; e < 8; ++e) pk[e] = (__bf16)(acc[s][j][e] + bv);
                *(v8bf*)(Vt + (bh * HD + d) * T_SEQ + t0b + s * 16) = pk;
            }
        }
    } else {
        __bf16* dst = (sel == 0) ? Qb : Kb;   // [BH][T][64]
        #pragma unroll
        for (int j = 0; j < 4; ++j) {
            const float bv = bias[N0 + j * 16 + mrow];
            const int d = j * 16 + mrow;
            #pragma unroll
            for (int s = 0; s < 2; ++s) {
                #pragma unroll
                for (int e = 0; e < 8; ++e) {
                    dst[(bh * T_SEQ + t0b + s * 16 + e) * HD + d] =
                        (__bf16)(acc[s][j][e] + bv);
                }
            }
        }
    }
}

// ---------------------------------------------------------------------------
// Kernel 2: flash attention.  grid=(T/64, BH), block=128 (4 independent waves).
// Wave owns 16 query rows; iterates causal 32-key blocks with online softmax.
// ---------------------------------------------------------------------------
__global__ __launch_bounds__(128)
void mha_flash_attn(const __bf16* __restrict__ Qb, const __bf16* __restrict__ Kb,
                    const __bf16* __restrict__ Vt, float* __restrict__ out) {
    __shared__ __bf16 sP[4 * 16 * LDSTRIDE];   // per-wave 16x32 P tile, padded

    const int lane = threadIdx.x & 31;
    const int wave = threadIdx.x >> 5;
    const int half = lane >> 4;
    const int mrow = lane & 15;
    const int base = half * 8;

    const int bh = blockIdx.y;
    const int q0 = blockIdx.x * 64 + wave * 16;
    __bf16* sPw = sP + wave * 16 * LDSTRIDE;

    // ---- Q fragments (A layout): d = 0..31 and 32..63 ----
    const __bf16* qr = Qb + (bh * T_SEQ + q0 + mrow) * HD;
    const v16bf aQ0 = load_frag(qr + base);
    const v16bf aQ1 = load_frag(qr + 32 + base);

    float mstat[8], lstat[8];
    v8f oAcc[4] = {};
    #pragma unroll
    for (int e = 0; e < 8; ++e) { mstat[e] = -__builtin_inff(); lstat[e] = 0.0f; }

    const int nblk = (q0 + 47) >> 5;     // causal: keys up to q0+15
    for (int kb = 0; kb < nblk; ++kb) {
        const int tk0 = kb * 32;

        // ---- K fragments (B layout: lane = key, elements = d) ----
        const __bf16* krA = Kb + (bh * T_SEQ + tk0 + mrow) * HD;
        const __bf16* krB = krA + 16 * HD;
        const v16bf bA0 = load_frag(krA + base);
        const v16bf bA1 = load_frag(krA + 32 + base);
        const v16bf bB0 = load_frag(krB + base);
        const v16bf bB1 = load_frag(krB + 32 + base);

        // ---- S = Q K^T  (16 x 32), fp32 accum over hd=64 ----
        v8f sA = {}, sB2 = {};
        sA  = wmma_bf16(aQ0, bA0, sA);
        sA  = wmma_bf16(aQ1, bA1, sA);
        sB2 = wmma_bf16(aQ0, bB0, sB2);
        sB2 = wmma_bf16(aQ1, bB1, sB2);

        // ---- mask (causal) then scale (reference order), online softmax ----
        const int kA = tk0 + mrow;
        const int kB = kA + 16;
        float fA[8], fB[8];
        #pragma unroll
        for (int e = 0; e < 8; ++e) {
            const int qrow = q0 + half * 8 + e;
            fA[e] = (kA <= qrow) ? sA[e]  * SCALE : -__builtin_inff();
            fB[e] = (kB <= qrow) ? sB2[e] * SCALE : -__builtin_inff();
        }
        #pragma unroll
        for (int e = 0; e < 8; ++e) {
            float rm = fmaxf(fA[e], fB[e]);
            rm = fmaxf(rm, __shfl_xor(rm, 1, 16));
            rm = fmaxf(rm, __shfl_xor(rm, 2, 16));
            rm = fmaxf(rm, __shfl_xor(rm, 4, 16));
            rm = fmaxf(rm, __shfl_xor(rm, 8, 16));
            const float mn   = fmaxf(mstat[e], rm);
            const float pA   = __expf(fA[e] - mn);
            const float pB   = __expf(fB[e] - mn);
            const float corr = __expf(mstat[e] - mn);
            float rs = pA + pB;
            rs += __shfl_xor(rs, 1, 16);
            rs += __shfl_xor(rs, 2, 16);
            rs += __shfl_xor(rs, 4, 16);
            rs += __shfl_xor(rs, 8, 16);
            lstat[e] = lstat[e] * corr + rs;
            mstat[e] = mn;
            #pragma unroll
            for (int j = 0; j < 4; ++j) oAcc[j][e] *= corr;
            // stash P (C layout -> LDS) for re-layout into A format
            const int row = half * 8 + e;
            sPw[row * LDSTRIDE + mrow]      = (__bf16)pA;
            sPw[row * LDSTRIDE + 16 + mrow] = (__bf16)pB;
        }
        LDS_FENCE();   // wave-internal DS ops are in-order; keep compiler honest

        // ---- reload P in A-fragment layout: 2x ds_load_b128 ----
        const v16bf aP = load_frag(&sPw[mrow * LDSTRIDE + base]);
        LDS_FENCE();

        // ---- O += P V  (V^T layout makes B fragments contiguous loads) ----
        #pragma unroll
        for (int j = 0; j < 4; ++j) {
            const __bf16* vr = Vt + (bh * HD + j * 16 + mrow) * T_SEQ + tk0;
            oAcc[j] = wmma_bf16(aP, load_frag(vr + base), oAcc[j]);
        }
    }

    // ---- normalize and store: out[b][t][h*64+d], fp32 ----
    const int bb = bh >> 4;
    const int h  = bh & 15;
    #pragma unroll
    for (int j = 0; j < 4; ++j) {
        const int d = j * 16 + mrow;
        #pragma unroll
        for (int e = 0; e < 8; ++e) {
            const int t = q0 + half * 8 + e;
            out[(bb * T_SEQ + t) * E_DIM + h * HD + d] = oAcc[j][e] / lstat[e];
        }
    }
}

// ---------------------------------------------------------------------------
extern "C" void kernel_launch(void* const* d_in, const int* in_sizes, int n_in,
                              void* d_out, int out_size, void* d_ws, size_t ws_size,
                              hipStream_t stream) {
    const float* x  = (const float*)d_in[0];   // [2, 2048, 1024]
    const float* W  = (const float*)d_in[1];   // [1024, 3072]
    const float* b  = (const float*)d_in[2];   // [3072]
    float* out      = (float*)d_out;           // [2, 2048, 1024]

    __bf16* xb = (__bf16*)d_ws;                        // [4096][1024]
    __bf16* Wt = xb + (size_t)M_TOT * E_DIM;           // [3072][1024]
    __bf16* Qb = Wt + (size_t)N3 * E_DIM;              // [32][2048][64]
    __bf16* Kb = Qb + (size_t)BH * T_SEQ * HD;         // [32][2048][64]
    __bf16* Vt = Kb + (size_t)BH * T_SEQ * HD;         // [32][64][2048]

    cvt_x_bf16<<<(M_TOT * E_DIM) / (256 * 8), 256, 0, stream>>>(x, xb);
    transpose_w_bf16<<<dim3(N3 / 32, E_DIM / 32), 256, 0, stream>>>(W, Wt);
    mha_qkv_gemm<<<dim3(N3 / 64, M_TOT / 128), 128, 0, stream>>>(
        xb, Wt, b, Qb, Kb, Vt);
    mha_flash_attn<<<dim3(T_SEQ / 64, BH), 128, 0, stream>>>(
        Qb, Kb, Vt, out);
}
